// CoAttentionLayer_66924180407196
// MI455X (gfx1250) — matrix operand order repaired and checked by reading, not compiled, not run on hardware
//
#include <hip/hip_runtime.h>

// ---------------------------------------------------------------------------
// Types for CDNA5 WMMA (f16 A/B, f32 accumulate)
// ---------------------------------------------------------------------------
typedef __attribute__((ext_vector_type(16))) _Float16 v16h;
typedef __attribute__((ext_vector_type(8)))  float    v8f;

union Frag {
    uint4 u[2];   // 32 bytes = 16 f16
    v16h  v;
};

static __device__ __forceinline__ v8f wmma_f16(v16h a, v16h b, v8f c) {
    // D = A(16x32) * B(32x16) + C(16x16), f32 accumulate
    return __builtin_amdgcn_wmma_f32_16x16x32_f16(
        /*neg_a=*/false, a, /*neg_b=*/false, b,
        /*c_mod=*/(short)0, c, /*reuse_a=*/false, /*reuse_b=*/false);
}

// pack two f32 -> f16x2 in one v_cvt_pk_rtz instruction
static __device__ __forceinline__ unsigned pk2h(float a, float b) {
    auto h = __builtin_amdgcn_cvt_pkrtz(a, b);   // __fp16 ext_vector(2)
    return __builtin_bit_cast(unsigned, h);
}

static __device__ __forceinline__ float fast_tanh(float x) {
#if __has_builtin(__builtin_amdgcn_tanhf)
    return __builtin_amdgcn_tanhf(x);
#elif __has_builtin(__builtin_amdgcn_tanh_f32)
    return __builtin_amdgcn_tanh_f32(x);
#else
    float r;
    asm volatile("v_tanh_f32 %0, %1\n\tv_nop" : "=v"(r) : "v"(x));
    return r;
#endif
}

// Problem constants
#define NB   128
#define LSEQ 512
#define DIM  256
#define ADIM 64

// ---------------------------------------------------------------------------
// K1: convert Q and K to f16 (vector-of-4 per thread)
// ---------------------------------------------------------------------------
__global__ __launch_bounds__(256) void k_convert_qk(
    const float* __restrict__ Q, const float* __restrict__ K,
    _Float16* __restrict__ Qh, _Float16* __restrict__ Kh, int n4)
{
    int i = blockIdx.x * 256 + threadIdx.x;
    if (i >= n4) return;
    float4 q = ((const float4*)Q)[i];
    float4 k = ((const float4*)K)[i];
    uint2 pq, pk;
    pq.x = pk2h(q.x, q.y);
    pq.y = pk2h(q.z, q.w);
    pk.x = pk2h(k.x, k.y);
    pk.y = pk2h(k.z, k.w);
    ((uint2*)Qh)[i] = pq;
    ((uint2*)Kh)[i] = pk;
}

// ---------------------------------------------------------------------------
// K2: transpose + convert weight matrices to f16
//   w_relT[o,i] = w_rel[i,o]  (256x256);  WqT/WkT[a,i] = W[i,a] (64x256)
// ---------------------------------------------------------------------------
__global__ __launch_bounds__(256) void k_convert_w(
    const float* __restrict__ wrel, const float* __restrict__ Wq,
    const float* __restrict__ Wk,
    _Float16* __restrict__ wrelT, _Float16* __restrict__ WqT,
    _Float16* __restrict__ WkT)
{
    int i = blockIdx.x * 256 + threadIdx.x;
    if (i < 256 * 256) {
        int r = i >> 8, c = i & 255;
        wrelT[c * 256 + r] = (_Float16)wrel[i];
    }
    if (i < 256 * 64) {
        int r = i >> 6, c = i & 63;
        WqT[c * 256 + r] = (_Float16)Wq[i];
        WkT[c * 256 + r] = (_Float16)Wk[i];
    }
}

// ---------------------------------------------------------------------------
// K3: row-block WMMA GEMM.
//   A rows from Ah [B,512,256] f16.
//   If QWout != null: QWout[b,q,:] = f16( A @ w_relT^T )   (N=256)
//   Always: projTout[b, a, q]  = f16( A @ WprojT^T )[q,a]  (N=64, transposed)
// Block = 256 threads = 8 waves, each wave owns 16 rows; grid = B*4.
// ---------------------------------------------------------------------------
__global__ __launch_bounds__(256) void k_gemm_qw_proj(
    const _Float16* __restrict__ Ah,
    const _Float16* __restrict__ WrelT,
    _Float16* __restrict__ QWout,
    const _Float16* __restrict__ WprojT,
    _Float16* __restrict__ projTout)
{
    __shared__ _Float16 ptile[8][ADIM * 16];   // 8 waves x 2KB = 16KB

    const int blk = blockIdx.x;
    const int b = blk >> 2;
    const int rbase = (blk & 3) * 128;
    const int t = threadIdx.x;
    const int wave = t >> 5, L = t & 31;
    const int Lm = L & 15, half = L >> 4;
    const int r0 = rbase + wave * 16;

    // Load the wave's 8 A-fragments (rows r0..r0+15, full 256 contraction dim)
    const _Float16* Arow = Ah + ((size_t)b * LSEQ + r0 + Lm) * DIM;
    Frag a[8];
#pragma unroll
    for (int s = 0; s < 8; s++) {
        const char* p = (const char*)Arow + s * 64 + (half ? 32 : 0);
        a[s].u[0] = *(const uint4*)p;
        a[s].u[1] = *(const uint4*)(p + 16);
    }

    // ---- QW = A @ w_rel  (N = 256 -> 16 n-tiles) ----
    if (QWout) {
#pragma unroll 1
        for (int nt = 0; nt < 16; nt++) {
            v8f c = {0.f, 0.f, 0.f, 0.f, 0.f, 0.f, 0.f, 0.f};
            const char* Brow = (const char*)(WrelT + (size_t)(nt * 16 + Lm) * DIM)
                               + (half ? 32 : 0);
#pragma unroll
            for (int s = 0; s < 8; s++) {
                Frag bf_;
                bf_.u[0] = *(const uint4*)(Brow + s * 64);
                bf_.u[1] = *(const uint4*)(Brow + s * 64 + 16);
                c = wmma_f16(a[s].v, bf_.v, c);
            }
            _Float16* out = QWout + ((size_t)b * LSEQ + r0) * DIM + nt * 16 + Lm;
#pragma unroll
            for (int r = 0; r < 8; r++)
                out[(size_t)(r + half * 8) * DIM] = (_Float16)c[r];
        }
    }

    // ---- proj = A @ Wproj (N = 64 -> 4 n-tiles), write transposed f16 ----
#pragma unroll 1
    for (int nt = 0; nt < 4; nt++) {
        v8f c = {0.f, 0.f, 0.f, 0.f, 0.f, 0.f, 0.f, 0.f};
        const char* Brow = (const char*)(WprojT + (size_t)(nt * 16 + Lm) * DIM)
                           + (half ? 32 : 0);
#pragma unroll
        for (int s = 0; s < 8; s++) {
            Frag bf_;
            bf_.u[0] = *(const uint4*)(Brow + s * 64);
            bf_.u[1] = *(const uint4*)(Brow + s * 64 + 16);
            c = wmma_f16(a[s].v, bf_.v, c);
        }
        const int acol = nt * 16 + Lm;
#pragma unroll
        for (int r = 0; r < 8; r++)
            ptile[wave][acol * 16 + r + half * 8] = (_Float16)c[r];
    }
    asm volatile("s_wait_dscnt 0" ::: "memory");
    __builtin_amdgcn_wave_barrier();

    // Coalesced write-out of the 64x16 transposed tile (as dwords)
    const unsigned int* lds32 = (const unsigned int*)&ptile[wave][0];
    _Float16* pT = projTout + (size_t)b * ADIM * LSEQ;
#pragma unroll
    for (int i = 0; i < 8; i++) {
        int idx = i * 32 + L;       // 0..255 dword index of the tile
        int arow = idx >> 3;        // 8 dwords (16 f16) per row
        int cc = idx & 7;
        *(unsigned int*)((char*)(pT + (size_t)arow * LSEQ + r0) + cc * 4) = lds32[idx];
    }
}

// ---------------------------------------------------------------------------
// K4: attention pass (launched twice, symmetric).
//   rel_chunk[16 x 32] = tanh( Arows @ Brows^T )      (contraction DIM=256)
//   S[16 x 64]        += rel_chunk @ projT^T          (contraction k)
//   logits[row]        = sum_a tanh(selfT[a,row] + S[row,a]) * Watt[a]
// Block = 256 threads = 8 waves x 16 rows; grid = B*4.
// ---------------------------------------------------------------------------
__global__ __launch_bounds__(256) void k_attn_pass(
    const _Float16* __restrict__ Ah,      // [B,512,256] rel A-source
    const _Float16* __restrict__ Bh,      // [B,512,256] rel B-source
    const _Float16* __restrict__ projT,   // [B,64,512] other-side proj
    const _Float16* __restrict__ selfT,   // [B,64,512] own proj (bias)
    const float* __restrict__ Watt,       // [64]
    float* __restrict__ logits)           // [B,512]
{
    __shared__ uint4 bstage[1024];                // 32 rows x 512B = 16KB
    __shared__ _Float16 reltile[8][16 * 32];      // 8 waves x 1KB

    const int blk = blockIdx.x;
    const int b = blk >> 2;
    const int rbase = (blk & 3) * 128;
    const int t = threadIdx.x;
    const int wave = t >> 5, L = t & 31;
    const int Lm = L & 15, half = L >> 4;
    const int r0 = rbase + wave * 16;

    // Wave-resident A fragments (16 rows x 256 contraction)
    const _Float16* Arow = Ah + ((size_t)b * LSEQ + r0 + Lm) * DIM;
    Frag a[8];
#pragma unroll
    for (int s = 0; s < 8; s++) {
        const char* p = (const char*)Arow + s * 64 + (half ? 32 : 0);
        a[s].u[0] = *(const uint4*)p;
        a[s].u[1] = *(const uint4*)(p + 16);
    }

    v8f S[4];
#pragma unroll
    for (int nt = 0; nt < 4; nt++)
        S[nt] = (v8f){0.f, 0.f, 0.f, 0.f, 0.f, 0.f, 0.f, 0.f};

    const _Float16* Bbase = Bh + (size_t)b * LSEQ * DIM;
    const _Float16* pT = projT + (size_t)b * ADIM * LSEQ;

#pragma unroll 1
    for (int c0 = 0; c0 < LSEQ; c0 += 32) {
        __syncthreads();
        // Cooperative stage of B rows c0..c0+31 (16KB)
        const uint4* g = (const uint4*)(Bbase + (size_t)c0 * DIM);
#pragma unroll
        for (int i = 0; i < 4; i++)
            bstage[i * 256 + t] = g[i * 256 + t];
        __syncthreads();

        // rel chunk: two 16x16 C tiles over contraction 256
#pragma unroll
        for (int nt2 = 0; nt2 < 2; nt2++) {
            v8f c2 = {0.f, 0.f, 0.f, 0.f, 0.f, 0.f, 0.f, 0.f};
            const char* brow = (const char*)bstage + (nt2 * 16 + Lm) * 512
                               + (half ? 32 : 0);
#pragma unroll
            for (int s = 0; s < 8; s++) {
                Frag bf_;
                bf_.u[0] = *(const uint4*)(brow + s * 64);
                bf_.u[1] = *(const uint4*)(brow + s * 64 + 16);
                c2 = wmma_f16(a[s].v, bf_.v, c2);
            }
            // tanh + repack (C layout -> A layout) through LDS
#pragma unroll
            for (int r = 0; r < 8; r++) {
                float v = fast_tanh(c2[r]);
                reltile[wave][(r + half * 8) * 32 + nt2 * 16 + Lm] = (_Float16)v;
            }
        }
        asm volatile("s_wait_dscnt 0" ::: "memory");
        __builtin_amdgcn_wave_barrier();

        Frag ra;
        {
            const char* rrow = (const char*)&reltile[wave][0] + Lm * 64
                               + (half ? 32 : 0);
            ra.u[0] = *(const uint4*)rrow;
            ra.u[1] = *(const uint4*)(rrow + 16);
        }
        // S += rel_chunk @ projT^T  (4 n-tiles over A dim)
#pragma unroll
        for (int nt = 0; nt < 4; nt++) {
            const char* prow = (const char*)(pT + (size_t)(nt * 16 + Lm) * LSEQ
                                             + c0 + (half ? 16 : 0));
            Frag bf_;
            bf_.u[0] = *(const uint4*)prow;
            bf_.u[1] = *(const uint4*)(prow + 16);
            S[nt] = wmma_f16(ra.v, bf_.v, S[nt]);
        }
    }

    // Epilogue: logits[row] = sum_a tanh(bias + S) * Watt[a]
    float part[8];
#pragma unroll
    for (int r = 0; r < 8; r++) part[r] = 0.f;

#pragma unroll
    for (int nt = 0; nt < 4; nt++) {
        const int acol = nt * 16 + Lm;
        const float w = Watt[acol];
        const _Float16* srow = selfT + (size_t)acol * LSEQ + r0;
#pragma unroll
        for (int r = 0; r < 8; r++) {
            float bias = (float)srow[r + half * 8];
            part[r] += fast_tanh(bias + S[nt][r]) * w;
        }
    }
    // reduce over the 16 lanes of each half-wave
#pragma unroll
    for (int m = 1; m < 16; m <<= 1) {
#pragma unroll
        for (int r = 0; r < 8; r++)
            part[r] += __shfl_xor(part[r], m, 32);
    }
    if (Lm == 0) {
        float* lrow = logits + (size_t)b * LSEQ + r0 + half * 8;
#pragma unroll
        for (int r = 0; r < 8; r++) lrow[r] = part[r];
    }
}

// ---------------------------------------------------------------------------
// K5: softmax over sequence dim (512) + scale rows.
// grid = B*4 ; each block redundantly computes the full softmax (cheap) and
// scales its own 128-row chunk for 4x the bandwidth concurrency.
// ---------------------------------------------------------------------------
__global__ __launch_bounds__(256) void k_softmax_scale(
    const float* __restrict__ logits, const float* __restrict__ src,
    float* __restrict__ outw, float* __restrict__ outatt)
{
    __shared__ float red[256];
    __shared__ float wl[LSEQ];
    const int blk = blockIdx.x;
    const int b = blk >> 2;
    const int chunk = blk & 3;
    const int t = threadIdx.x;

    const float* lg = logits + (size_t)b * LSEQ;
    float l0 = lg[t], l1 = lg[t + 256];
    red[t] = fmaxf(l0, l1);
    __syncthreads();
    for (int s = 128; s > 0; s >>= 1) {
        if (t < s) red[t] = fmaxf(red[t], red[t + s]);
        __syncthreads();
    }
    float mx = red[0];
    __syncthreads();
    float e0 = __expf(l0 - mx), e1 = __expf(l1 - mx);
    red[t] = e0 + e1;
    __syncthreads();
    for (int s = 128; s > 0; s >>= 1) {
        if (t < s) red[t] += red[t + s];
        __syncthreads();
    }
    float inv = 1.0f / red[0];
    wl[t] = e0 * inv;
    wl[t + 256] = e1 * inv;
    __syncthreads();

    const int row0 = chunk * 128;
    if (t < 128)
        outatt[(size_t)b * LSEQ + row0 + t] = wl[row0 + t];

    const float4* s4 = (const float4*)(src + ((size_t)b * LSEQ + row0) * DIM);
    float4* o4 = (float4*)(outw + ((size_t)b * LSEQ + row0) * DIM);
    for (int i = t; i < 128 * (DIM / 4); i += 256) {
        int row = i >> 6;   // 64 float4 per row
        float4 v = s4[i];
        float w = wl[row0 + row];
        v.x *= w; v.y *= w; v.z *= w; v.w *= w;
        o4[i] = v;
    }
}

// ---------------------------------------------------------------------------
// Host launcher
// ---------------------------------------------------------------------------
extern "C" void kernel_launch(void* const* d_in, const int* in_sizes, int n_in,
                              void* d_out, int out_size, void* d_ws, size_t ws_size,
                              hipStream_t stream) {
    const float* Q    = (const float*)d_in[0];
    const float* K    = (const float*)d_in[1];
    const float* wrel = (const float*)d_in[2];
    const float* Wq   = (const float*)d_in[3];
    const float* Wk   = (const float*)d_in[4];
    const float* Wqa  = (const float*)d_in[5];
    const float* Wka  = (const float*)d_in[6];
    float* out = (float*)d_out;

    // Workspace layout (bytes, 256-aligned)
    char* ws = (char*)d_ws;
    const size_t szQK   = (size_t)NB * LSEQ * DIM * 2;     // 33,554,432
    const size_t szProj = (size_t)NB * ADIM * LSEQ * 2;    //  8,388,608
    _Float16* Qh      = (_Float16*)(ws);
    _Float16* Kh      = (_Float16*)(ws + szQK);
    _Float16* QWh     = (_Float16*)(ws + 2 * szQK);
    _Float16* qprojT  = (_Float16*)(ws + 3 * szQK);
    _Float16* kprojT  = (_Float16*)(ws + 3 * szQK + szProj);
    _Float16* wrelT   = (_Float16*)(ws + 3 * szQK + 2 * szProj);
    _Float16* WqT     = (_Float16*)(ws + 3 * szQK + 2 * szProj + 131072);
    _Float16* WkT     = (_Float16*)(ws + 3 * szQK + 2 * szProj + 131072 + 32768);
    float* logits_q = (float*)(ws + 3 * szQK + 2 * szProj + 131072 + 65536);
    float* logits_k = (float*)(ws + 3 * szQK + 2 * szProj + 131072 + 65536 + 262144);

    const int n4 = NB * LSEQ * DIM / 4;
    k_convert_qk<<<n4 / 256, 256, 0, stream>>>(Q, K, Qh, Kh, n4);
    k_convert_w<<<256, 256, 0, stream>>>(wrel, Wq, Wk, wrelT, WqT, WkT);

    // Q rows: QWh + qprojT ; K rows: kprojT only
    k_gemm_qw_proj<<<NB * 4, 256, 0, stream>>>(Qh, wrelT, QWh, WqT, qprojT);
    k_gemm_qw_proj<<<NB * 4, 256, 0, stream>>>(Kh, nullptr, nullptr, WkT, kprojT);

    // Pass A (rows = q): rel = tanh(QW @ K^T); S_q = rel @ k_proj
    k_attn_pass<<<NB * 4, 256, 0, stream>>>(QWh, Kh, kprojT, qprojT, Wqa, logits_q);
    // Pass B (rows = k): rel^T = tanh(K @ QW^T); S_k = rel^T @ q_proj
    k_attn_pass<<<NB * 4, 256, 0, stream>>>(Kh, QWh, qprojT, kprojT, Wka, logits_k);

    float* out_wq  = out;                                   // [B,512,256]
    float* out_wk  = out + (size_t)NB * LSEQ * DIM;         // [B,512,256]
    float* out_qa  = out + (size_t)2 * NB * LSEQ * DIM;     // [B,512,1]
    float* out_ka  = out_qa + (size_t)NB * LSEQ;            // [B,512,1]

    k_softmax_scale<<<NB * 4, 256, 0, stream>>>(logits_q, Q, out_wq, out_qa);
    k_softmax_scale<<<NB * 4, 256, 0, stream>>>(logits_k, K, out_wk, out_ka);
}
